// CausalSelfAttention_47794396070537
// MI455X (gfx1250) — matrix operand order repaired and checked by reading, not compiled, not run on hardware
//
#include <hip/hip_runtime.h>
#include <hip/hip_bf16.h>

// ---------------------------------------------------------------------------
// CDNA5 (gfx1250): all four matmul stages on V_WMMA_F32_16X16X32_F16.
// Operands are pre-converted once (x -> f16, weights -> f16 transposed) so
// every WMMA fragment in the GEMMs is exactly two global_load_b128 per lane;
// flash attention stages tiles in LDS with every fragment read a pair of
// 16B-aligned ds_load_b128. wave32 throughout.
// ---------------------------------------------------------------------------

typedef __attribute__((ext_vector_type(16))) _Float16 v16h;
typedef __attribute__((ext_vector_type(8)))  float    v8f;

#define BB    2
#define SS    2048
#define DD    1024
#define NH    16
#define MH    4
#define HDIM  64
#define GIN_  12
#define ROWS  (BB*SS)      // 4096
#define KVLD  (MH*HDIM)    // 256

union FragH { v16h v; _Float16 h[16]; unsigned short s[16]; unsigned int w[8]; };
union Acc8  { v8f  v; float e[8]; };

__device__ __forceinline__ unsigned short f2h(float f) {
  union { _Float16 h; unsigned short s; } u;
  u.h = (_Float16)f;
  return u.s;
}

// Build a fragment from two 16-byte chunks (8 consecutive halves each).
// A and B fragments use contiguous per-lane K (k = hf*16 + s): chunks at
// hf*16 and hf*16+8.
__device__ __forceinline__ v16h frag_from2(const unsigned short* p0,
                                           const unsigned short* p1) {
  FragH u;
  uint4 a = *(const uint4*)p0;
  uint4 b = *(const uint4*)p1;
  u.w[0]=a.x; u.w[1]=a.y; u.w[2]=a.z; u.w[3]=a.w;
  u.w[4]=b.x; u.w[5]=b.y; u.w[6]=b.z; u.w[7]=b.w;
  return u.v;
}

// ---------------------------------------------------------------------------
// Prep kernels: f32 -> f16 copy, and f32 -> f16 transpose (WT[c][k] = W[k][c]).
// ---------------------------------------------------------------------------
__global__ void __launch_bounds__(256)
k_f2h_copy(const float* __restrict__ src, unsigned short* __restrict__ dst, int n) {
  int i = blockIdx.x*256 + threadIdx.x;
  if (i < n) dst[i] = f2h(src[i]);
}

__global__ void __launch_bounds__(256)
k_w2h_t(const float* __restrict__ W, unsigned short* __restrict__ WT, int K, int C) {
  int idx = blockIdx.x*256 + threadIdx.x;
  if (idx >= K*C) return;
  int k = idx / C, c = idx % C;            // coalesced read along c
  WT[(size_t)c*K + k] = f2h(W[(size_t)k*C + c]);
}

// ---------------------------------------------------------------------------
// Kernel 1: QKV projections, no LDS. Each wave: 16x64 output strip.
// A from xh (f16 row-major), B from WT (f16 transposed): all b128 loads.
// grid = (256 row tiles, 24 col groups): 0..15 q, 16..19 k, 20..23 v.
// ---------------------------------------------------------------------------
__global__ void __launch_bounds__(32)
k_qkv_gemm(const unsigned short* __restrict__ xh,
           const unsigned short* __restrict__ wqT,
           const unsigned short* __restrict__ wkT,
           const unsigned short* __restrict__ wvT,
           float* __restrict__ qraw, float* __restrict__ kraw,
           float* __restrict__ vraw) {
  int lane = threadIdx.x;
  int col = lane & 15, hf = lane >> 4;
  int rt = blockIdx.x, ctg = blockIdx.y;
  const unsigned short* WT; float* out; int ldo, c0;
  if (ctg < 16)      { WT = wqT; out = qraw; ldo = DD;   c0 = ctg*64; }
  else if (ctg < 20) { WT = wkT; out = kraw; ldo = KVLD; c0 = (ctg-16)*64; }
  else               { WT = wvT; out = vraw; ldo = KVLD; c0 = (ctg-20)*64; }
  int r0 = rt*16;
  Acc8 acc[4];
#pragma unroll
  for (int t = 0; t < 4; ++t)
#pragma unroll
    for (int i = 0; i < 8; ++i) acc[t].e[i] = 0.f;

  const unsigned short* arow = xh + (size_t)(r0 + col)*DD;   // A row, this lane
  const unsigned short* bcol = WT + (size_t)(c0 + col)*DD;   // B col, this lane

  for (int kk = 0; kk < DD; kk += 32) {
    __builtin_prefetch(arow + kk + 512, 0, 1);
    v16h a = frag_from2(arow + kk + hf*16, arow + kk + hf*16 + 8);
#pragma unroll
    for (int t = 0; t < 4; ++t) {
      const unsigned short* pb = bcol + (size_t)(t*16)*DD + kk + hf*16;
      v16h bb = frag_from2(pb, pb + 8);
      acc[t].v = __builtin_amdgcn_wmma_f32_16x16x32_f16(false, a, false, bb,
                                                        (short)0, acc[t].v, false, false);
    }
  }
#pragma unroll
  for (int t = 0; t < 4; ++t)
#pragma unroll
    for (int r = 0; r < 8; ++r)
      out[(size_t)(r0 + r + 8*hf)*ldo + c0 + t*16 + col] = acc[t].e[r];
}

// ---------------------------------------------------------------------------
// Kernel 2: per-row epilogue: gates, RMS-norm + RoPE on q/k, ve mix on v.
// grid = 4096, 64 threads.
// ---------------------------------------------------------------------------
__device__ __forceinline__ void norm_rope(const float* __restrict__ src,
                                          unsigned short* __restrict__ dst, int s) {
  float v[HDIM]; float ssum = 0.f;
  for (int j = 0; j < HDIM; ++j) { v[j] = src[j]; ssum += v[j]*v[j]; }
  float rs = rsqrtf(ssum * (1.f/HDIM) + 1e-6f);
  float o[HDIM];
  for (int j = 0; j < 16; ++j) {
    float x1 = v[j]*rs, x2 = v[16+j]*rs;
    float fr = exp2f(-0.625f * (float)j);   // 1024^(-j/16) = 2^(-0.625 j)
    float ang = (float)s * fr;
    float cs = cosf(ang), sn = sinf(ang);
    o[j]    = x1*cs - x2*sn;
    o[16+j] = x2*cs + x1*sn;
  }
  for (int j = 32; j < HDIM; ++j) o[j] = v[j]*rs;
  for (int j = 0; j < HDIM; ++j) dst[j] = f2h(o[j]);
}

__global__ void __launch_bounds__(64)
k_epilogue(const float* __restrict__ x, const int* __restrict__ tok,
           const float* __restrict__ ve_embed,
           const float* __restrict__ vlam, const float* __restrict__ velam,
           const float* __restrict__ vegate, const float* __restrict__ agw,
           const float* __restrict__ qraw, const float* __restrict__ kraw,
           const float* __restrict__ vraw,
           unsigned short* __restrict__ qh, unsigned short* __restrict__ kh,
           unsigned short* __restrict__ vh, float* __restrict__ agate) {
  int row = blockIdx.x;
  int s = row % SS;
  int t = threadIdx.x;
  __shared__ float xg[GIN_];
  __shared__ float sveg[MH];
  if (t < GIN_) xg[t] = x[(size_t)row*DD + t];
  __syncthreads();
  if (t < NH) {
    float d = 0.f;
    for (int j = 0; j < GIN_; ++j) d += xg[j]*agw[j*NH + t];
    agate[row*NH + t] = 2.f / (1.f + __expf(-d));
  } else if (t >= 32 && t < 32+MH) {
    int m = t - 32; float d = 0.f;
    for (int j = 0; j < GIN_; ++j) d += xg[j]*vegate[j*MH + m];
    sveg[m] = 2.f / (1.f + __expf(-d));
  }
  __syncthreads();
  if (t < NH) {                                           // q heads
    norm_rope(qraw + (size_t)row*DD + t*HDIM, qh + (size_t)row*DD + t*HDIM, s);
  } else if (t >= 16 && t < 16+MH) {                      // k heads
    int m = t - 16;
    norm_rope(kraw + (size_t)row*KVLD + m*HDIM, kh + (size_t)row*KVLD + m*HDIM, s);
  } else if (t >= 32) {                                   // v mix: 32 thr x 8 elems
    int e = t - 32;
    int token = tok[row];
    float vl = vlam[0], vel = velam[0];
    for (int i = 0; i < 8; ++i) {
      int idx = e*8 + i; int m = idx >> 6;
      float vv = vl * vraw[(size_t)row*KVLD + idx]
               + vel * sveg[m] * ve_embed[(size_t)token*KVLD + idx];
      vh[(size_t)row*KVLD + idx] = f2h(vv);
    }
  }
}

// ---------------------------------------------------------------------------
// Kernel 3: flash attention. grid = (S/16 query tiles, B*NH heads), 32 thr.
// Q/K row-major, V transposed in LDS: every fragment read = 2x ds_load_b128.
// ---------------------------------------------------------------------------
__global__ void __launch_bounds__(32)
k_flash(const unsigned short* __restrict__ qh, const unsigned short* __restrict__ kh,
        const unsigned short* __restrict__ vh, const float* __restrict__ agate,
        unsigned short* __restrict__ attn) {
  __shared__ unsigned short Qs[16][72];   // 144B row stride (16B multiple)
  __shared__ unsigned short Ks[32][72];
  __shared__ unsigned short Vt[64][40];   // transposed: Vt[dim][key], 80B stride
  __shared__ unsigned short Ps[16][40];
  __shared__ float mI[16], lI[16], fI[16];
  int lane = threadIdx.x;
  int qt = blockIdx.x;
  int bh = blockIdx.y;          // 0..31
  int b = bh >> 4;
  int head = bh & 15;
  int m = head >> 2;            // GQA: head n attends kv-head n/G
  int col = lane & 15, hf = lane >> 4;

  const unsigned short* qbase = qh + (size_t)(b*SS + qt*16)*DD + head*HDIM;
  for (int i = 0; i < 4; ++i) {                     // 16x64 Q tile as uint4
    int e = lane + i*32; int r = e >> 3, c8 = (e & 7)*8;
    *(uint4*)&Qs[r][c8] = *(const uint4*)(qbase + (size_t)r*DD + c8);
  }
  if (lane < 16) { mI[lane] = -3e38f; lI[lane] = 0.f; }
  Acc8 acc[4];
#pragma unroll
  for (int t2 = 0; t2 < 4; ++t2)
#pragma unroll
    for (int i = 0; i < 8; ++i) acc[t2].e[i] = 0.f;

  int nc = qt/2 + 1;                                // causal chunk count
  for (int kc = 0; kc < nc; ++kc) {
    int kb = kc*32;
    const unsigned short* kbase = kh + (size_t)(b*SS + kb)*KVLD + m*HDIM;
    const unsigned short* vbase = vh + (size_t)(b*SS + kb)*KVLD + m*HDIM;
    __builtin_prefetch(kbase + 32*KVLD, 0, 1);
    __builtin_prefetch(vbase + 32*KVLD, 0, 1);
    __syncthreads();
    for (int i = 0; i < 8; ++i) {                   // 32x64 K tile as uint4
      int e = lane + i*32; int r = e >> 3, c8 = (e & 7)*8;
      *(uint4*)&Ks[r][c8] = *(const uint4*)(kbase + (size_t)r*KVLD + c8);
      uint4 vv = *(const uint4*)(vbase + (size_t)r*KVLD + c8);   // V: transpose-store
      union { uint4 q; unsigned short s[8]; } uu; uu.q = vv;
#pragma unroll
      for (int j = 0; j < 8; ++j) Vt[c8 + j][r] = uu.s[j];
    }
    __syncthreads();

    Acc8 sc0, sc1;
#pragma unroll
    for (int i = 0; i < 8; ++i) { sc0.e[i] = 0.f; sc1.e[i] = 0.f; }
#pragma unroll
    for (int kkc = 0; kkc < 2; ++kkc) {             // HDIM=64 in two K=32 steps
      v16h a = frag_from2(&Qs[col][kkc*32 + hf*16], &Qs[col][kkc*32 + hf*16 + 8]);
      // scores = Q @ K^T : B[k][n] = Ks[c*16+n][kkc*32+k] -> contiguous per lane
      v16h b0 = frag_from2(&Ks[col][kkc*32 + hf*16],    &Ks[col][kkc*32 + hf*16 + 8]);
      v16h b1 = frag_from2(&Ks[16+col][kkc*32 + hf*16], &Ks[16+col][kkc*32 + hf*16 + 8]);
      sc0.v = __builtin_amdgcn_wmma_f32_16x16x32_f16(false, a, false, b0,
                                                     (short)0, sc0.v, false, false);
      sc1.v = __builtin_amdgcn_wmma_f32_16x16x32_f16(false, a, false, b1,
                                                     (short)0, sc1.v, false, false);
    }

    bool needMask = (kb + 31) > (qt*16);
#pragma unroll
    for (int r = 0; r < 8; ++r) {
      int rr = r + 8*hf;
      int rowg = qt*16 + rr;
      float v0 = sc0.e[r] * 0.125f;                 // 1/sqrt(64)
      float v1 = sc1.e[r] * 0.125f;
      if (needMask) {
        if (kb + col      > rowg) v0 = -1e30f;
        if (kb + 16 + col > rowg) v1 = -1e30f;
      }
      float vmax = fmaxf(v0, v1);
#pragma unroll
      for (int off = 1; off <= 8; off <<= 1)        // row reduce, 16-lane group
        vmax = fmaxf(vmax, __shfl_xor(vmax, off));
      float mold = mI[rr];
      float mnew = fmaxf(mold, vmax);
      float p0 = __expf(v0 - mnew);
      float p1 = __expf(v1 - mnew);
      float rsum = p0 + p1;
#pragma unroll
      for (int off = 1; off <= 8; off <<= 1)
        rsum += __shfl_xor(rsum, off);
      float fac = __expf(mold - mnew);
      if (col == 0) { mI[rr] = mnew; fI[rr] = fac; lI[rr] = fac*lI[rr] + rsum; }
      Ps[rr][col]      = f2h(p0);                   // restripe P to A layout
      Ps[rr][16 + col] = f2h(p1);
    }
    __syncthreads();

    v16h pa = frag_from2(&Ps[col][hf*16], &Ps[col][hf*16 + 8]);
#pragma unroll
    for (int t2 = 0; t2 < 4; ++t2) {
#pragma unroll
      for (int r = 0; r < 8; ++r) acc[t2].e[r] *= fI[r + 8*hf];
      // B[k][n] = V[key=k][dim=t2*16+n] = Vt[t2*16+n][k] -> contiguous per lane
      v16h bv = frag_from2(&Vt[t2*16 + col][hf*16], &Vt[t2*16 + col][hf*16 + 8]);
      acc[t2].v = __builtin_amdgcn_wmma_f32_16x16x32_f16(false, pa, false, bv,
                                                         (short)0, acc[t2].v, false, false);
    }
  }
  __syncthreads();
#pragma unroll
  for (int t2 = 0; t2 < 4; ++t2) {
#pragma unroll
    for (int r = 0; r < 8; ++r) {
      int rr = r + 8*hf;
      int rowg = b*SS + qt*16 + rr;
      float g = agate[rowg*NH + head];
      float val = acc[t2].e[r] / lI[rr] * g;
      attn[(size_t)rowg*DD + head*HDIM + t2*16 + col] = f2h(val);
    }
  }
}

// ---------------------------------------------------------------------------
// Kernel 4: output projection attn(f16) @ woT(f16) -> f32, no LDS.
// grid = (256 row tiles, 64 col groups of 64), 32 thr.
// ---------------------------------------------------------------------------
__global__ void __launch_bounds__(32)
k_out_gemm(const unsigned short* __restrict__ attn,
           const unsigned short* __restrict__ woT,
           float* __restrict__ out) {
  int lane = threadIdx.x;
  int col = lane & 15, hf = lane >> 4;
  int r0 = blockIdx.x*16, c0 = blockIdx.y*64;
  Acc8 acc[4];
#pragma unroll
  for (int t = 0; t < 4; ++t)
#pragma unroll
    for (int i = 0; i < 8; ++i) acc[t].e[i] = 0.f;

  const unsigned short* arow = attn + (size_t)(r0 + col)*DD;
  const unsigned short* bcol = woT  + (size_t)(c0 + col)*DD;

  for (int kk = 0; kk < DD; kk += 32) {
    __builtin_prefetch(arow + kk + 512, 0, 1);
    v16h a = frag_from2(arow + kk + hf*16, arow + kk + hf*16 + 8);
#pragma unroll
    for (int t = 0; t < 4; ++t) {
      const unsigned short* pb = bcol + (size_t)(t*16)*DD + kk + hf*16;
      v16h bb = frag_from2(pb, pb + 8);
      acc[t].v = __builtin_amdgcn_wmma_f32_16x16x32_f16(false, a, false, bb,
                                                        (short)0, acc[t].v, false, false);
    }
  }
#pragma unroll
  for (int t = 0; t < 4; ++t)
#pragma unroll
    for (int r = 0; r < 8; ++r)
      out[(size_t)(r0 + r + 8*hf)*DD + c0 + t*16 + col] = acc[t].e[r];
}

// ---------------------------------------------------------------------------
extern "C" void kernel_launch(void* const* d_in, const int* in_sizes, int n_in,
                              void* d_out, int out_size, void* d_ws, size_t ws_size,
                              hipStream_t stream) {
  const float* x      = (const float*)d_in[0];
  const int*   tok    = (const int*)  d_in[1];
  // d_in[2] = mask (unused: causal structure is hard-coded)
  const float* wq     = (const float*)d_in[3];
  const float* wk     = (const float*)d_in[4];
  const float* wv     = (const float*)d_in[5];
  const float* wo     = (const float*)d_in[6];
  const float* ve     = (const float*)d_in[7];
  const float* vlam   = (const float*)d_in[8];
  const float* velam  = (const float*)d_in[9];
  const float* vegate = (const float*)d_in[10];
  const float* agw    = (const float*)d_in[11];

  const size_t R = (size_t)ROWS;
  float* qraw = (float*)d_ws;                              // 4096*1024 f32
  float* kraw = qraw + R*DD;                               // 4096*256  f32
  float* vraw = kraw + R*KVLD;                             // 4096*256  f32
  unsigned short* qh   = (unsigned short*)(vraw + R*KVLD); // f16 buffers
  unsigned short* kh   = qh + R*DD;
  unsigned short* vh   = kh + R*KVLD;
  unsigned short* attn = vh + R*KVLD;
  unsigned short* xh   = attn + R*DD;                      // x as f16
  unsigned short* wqT  = xh + R*DD;                        // transposed f16 weights
  unsigned short* wkT  = wqT + (size_t)DD*DD;
  unsigned short* wvT  = wkT + (size_t)KVLD*DD;
  unsigned short* woT  = wvT + (size_t)KVLD*DD;
  float* agate = (float*)(woT + (size_t)DD*DD);            // 4096*16 f32

  // one-time operand prep (f16 copy / transposed f16 weights)
  k_f2h_copy<<<(int)(R*DD/256), 256, 0, stream>>>(x, xh, (int)(R*DD));
  k_w2h_t<<<DD*DD/256,   256, 0, stream>>>(wq, wqT, DD, DD);
  k_w2h_t<<<DD*KVLD/256, 256, 0, stream>>>(wk, wkT, DD, KVLD);
  k_w2h_t<<<DD*KVLD/256, 256, 0, stream>>>(wv, wvT, DD, KVLD);
  k_w2h_t<<<DD*DD/256,   256, 0, stream>>>(wo, woT, DD, DD);

  k_qkv_gemm<<<dim3(ROWS/16, 24), 32, 0, stream>>>(xh, wqT, wkT, wvT,
                                                   qraw, kraw, vraw);
  k_epilogue<<<ROWS, 64, 0, stream>>>(x, tok, ve, vlam, velam, vegate, agw,
                                      qraw, kraw, vraw, qh, kh, vh, agate);
  k_flash<<<dim3(SS/16, BB*NH), 32, 0, stream>>>(qh, kh, vh, agate, attn);
  k_out_gemm<<<dim3(ROWS/16, DD/16), 32, 0, stream>>>(attn, woT, (float*)d_out);
}
